// CDMamba_79224966742826
// MI455X (gfx1250) — matrix operand order repaired and checked by reading, not compiled, not run on hardware
//
#include <hip/hip_runtime.h>
#include <hip/hip_bf16.h>
#include <math.h>

#define BSZ     4
#define LSEQ    4096
#define DMODEL  512
#define DINNER  1024
#define DTRANK  32
#define DSTATE  16
#define NROWS   (BSZ * LSEQ)            // 16384 rows (B*L)
#define XPN     (DTRANK + 2 * DSTATE)   // 64 = xproj output dim
#define NCH     64                      // scan chunks
#define TC      64                      // timesteps per chunk (NCH*TC == LSEQ)

typedef __bf16 bf16_t;
typedef __attribute__((ext_vector_type(16))) __bf16 v16bf;
typedef __attribute__((ext_vector_type(8)))  float  v8f;

struct alignas(16) U128 { unsigned int x, y, z, w; };

// ---------------------------------------------------------------------------
// WMMA fragment loads (contiguous k order shared by A and B, wave32)
// A: M x K row-major bf16.  lane<16: elems[0:16]=K k..k+15
//                           lane>=16: elems[0:16]=K k+16..k+31.
__device__ __forceinline__ v16bf load_frag_a(const bf16_t* A, int lda, int row,
                                             int k, int lane) {
  const int kb = k + ((lane >> 4) << 4);
  union { v16bf v; U128 q[2]; } u;
  const bf16_t* p = A + (size_t)row * lda + kb;
  u.q[0] = *reinterpret_cast<const U128*>(p);
  u.q[1] = *reinterpret_cast<const U128*>(p + 8);
  return u.v;
}
// B: stored as weight matrix N x K row-major (i.e. B^T), so a column of B is a
// contiguous row here. lane = N index (mod 16); lanes 0-15 hold K k..k+15,
// lanes 16-31 hold K k+16..k+31.
__device__ __forceinline__ v16bf load_frag_b(const bf16_t* Bt, int ldb, int col,
                                             int k, int lane) {
  const int kk = k + ((lane >> 4) << 4);
  union { v16bf v; U128 q[2]; } u;
  const bf16_t* p = Bt + (size_t)col * ldb + kk;
  u.q[0] = *reinterpret_cast<const U128*>(p);
  u.q[1] = *reinterpret_cast<const U128*>(p + 8);
  return u.v;
}

// ---------------------------------------------------------------------------
// C[M,N] (f32) = A[M,K](bf16, lda) * Bt[N,K](bf16, ldb)^T, ldc = N.
// Block = 256 threads = 8 waves arranged 4(M) x 2(N).
// Wave tile 64x32 (4x2 WMMA tiles) -> block tile 256x64.
// M multiple of 256, N multiple of 64, K multiple of 32 (all true here).
// epi==1: softplus(x + bias[col]).
__global__ __launch_bounds__(256) void gemm_wmma_bf16(
    const bf16_t* __restrict__ A, int lda,
    const bf16_t* __restrict__ Bt, int ldb,
    float* __restrict__ C, int ldc, int K,
    const float* __restrict__ bias, int epi) {
  const int tid  = threadIdx.x;
  const int lane = tid & 31;
  const int wave = tid >> 5;
  const int mbase = blockIdx.y * 256 + (wave >> 1) * 64;
  const int nbase = blockIdx.x * 64  + (wave & 1) * 32;

  v8f acc[4][2];
#pragma unroll
  for (int i = 0; i < 4; ++i)
#pragma unroll
    for (int j = 0; j < 2; ++j) acc[i][j] = (v8f){};

  const int ar = mbase + (lane & 15);
  const int bc = nbase + (lane & 15);

  for (int k = 0; k < K; k += 32) {
    if (k + 32 < K) {
      __builtin_prefetch(A  + (size_t)ar * lda + k + 32, 0, 3);
      __builtin_prefetch(Bt + (size_t)bc * ldb + k + 32, 0, 3);
    }
    v16bf a[4], b[2];
#pragma unroll
    for (int i = 0; i < 4; ++i) a[i] = load_frag_a(A, lda, ar + i * 16, k, lane);
#pragma unroll
    for (int j = 0; j < 2; ++j) b[j] = load_frag_b(Bt, ldb, bc + j * 16, k, lane);
#pragma unroll
    for (int i = 0; i < 4; ++i)
#pragma unroll
      for (int j = 0; j < 2; ++j)
        acc[i][j] = __builtin_amdgcn_wmma_f32_16x16x32_bf16(
            false, a[i], false, b[j], (short)0, acc[i][j], false, false);
  }

  // D layout: VGPR r -> row r (lanes 0-15) / row 8+r (lanes 16-31); col = lane%16
  const int rb = (lane & 16) ? 8 : 0;
  const int cl = lane & 15;
#pragma unroll
  for (int i = 0; i < 4; ++i)
#pragma unroll
    for (int j = 0; j < 2; ++j) {
      const int col = nbase + j * 16 + cl;
      const float bv = (epi == 1) ? bias[col] : 0.0f;
#pragma unroll
      for (int r = 0; r < 8; ++r) {
        const int row = mbase + i * 16 + rb + r;
        float v = acc[i][j][r];
        if (epi == 1) {
          v += bv;
          v = (v > 20.0f) ? v : log1pf(__expf(v));   // softplus
        }
        C[(size_t)row * ldc + col] = v;
      }
    }
}

// ---------------------------------------------------------------------------
__global__ __launch_bounds__(256) void f32_to_bf16_kernel(
    const float* __restrict__ s, bf16_t* __restrict__ d, int n) {
  int i = blockIdx.x * blockDim.x + threadIdx.x;
  if (i < n) d[i] = (bf16_t)s[i];
}

// take first `cols` columns of src (row stride lds) into bf16 dst (row stride ldd)
__global__ __launch_bounds__(256) void slice_to_bf16_kernel(
    const float* __restrict__ src, int lds, bf16_t* __restrict__ dst, int ldd,
    int cols, int total) {
  int i = blockIdx.x * blockDim.x + threadIdx.x;
  if (i >= total) return;
  int r = i / cols, c = i - r * cols;
  dst[(size_t)r * ldd + c] = (bf16_t)src[(size_t)r * lds + c];
}

// ---------------------------------------------------------------------------
// Depthwise causal conv(4) + SiLU over x-half of xz [B,L,2*DINNER].
// reverse==1: taps look forward in original time (== causal conv on flipped seq).
__global__ __launch_bounds__(256) void conv_silu_kernel(
    const float* __restrict__ xz, const float* __restrict__ w,
    const float* __restrict__ cb, float* __restrict__ xc,
    bf16_t* __restrict__ xcb, int reverse) {
  int i = blockIdx.x * blockDim.x + threadIdx.x;      // over B*L*DINNER
  const int d = i & (DINNER - 1);
  const int t = (i / DINNER) & (LSEQ - 1);
  const int b = i / (DINNER * LSEQ);
  const float* xp = xz + (size_t)b * LSEQ * (2 * DINNER) + d;
  float acc = cb[d];
  if (!reverse) {
#pragma unroll
    for (int j = 0; j < 4; ++j) {
      int tt = t - 3 + j;
      if (tt >= 0) acc += w[d * 4 + j] * xp[(size_t)tt * (2 * DINNER)];
    }
  } else {
#pragma unroll
    for (int j = 0; j < 4; ++j) {
      int tt = t + 3 - j;
      if (tt < LSEQ) acc += w[d * 4 + j] * xp[(size_t)tt * (2 * DINNER)];
    }
  }
  float s = acc / (1.0f + __expf(-acc));
  xc[i] = s;
  xcb[i] = (bf16_t)s;
}

// ---------------------------------------------------------------------------
// Chunked selective scan: h[n] <- exp(dt*a[n])*h[n] + dt*x*B[n];  y = h.C
// Pass 1: per (b,chunk,d) run chunk with h0=0; emit chunk sum of dt and end-state.
__global__ __launch_bounds__(256) void scan_part1(
    const float* __restrict__ dt, const float* __restrict__ xc,
    const float* __restrict__ dbl, const float* __restrict__ Alog,
    float* __restrict__ carryS, float* __restrict__ sdt, int reverse) {
  const int gid = blockIdx.x * blockDim.x + threadIdx.x;  // B*NCH*DINNER
  const int d = gid & (DINNER - 1);
  const int c = (gid / DINNER) & (NCH - 1);
  const int b = gid / (DINNER * NCH);
  float a[DSTATE], h[DSTATE];
#pragma unroll
  for (int n = 0; n < DSTATE; ++n) { a[n] = -__expf(Alog[d * DSTATE + n]); h[n] = 0.0f; }
  float sum = 0.0f;
  for (int s = 0; s < TC; ++s) {
    const int sg = c * TC + s;
    const int t  = reverse ? (LSEQ - 1 - sg) : sg;
    const size_t base = (size_t)b * LSEQ + t;
    const float dt_t = dt[base * DINNER + d];
    const float dx   = dt_t * xc[base * DINNER + d];
    const float* bcp = dbl + base * XPN;
    sum += dt_t;
#pragma unroll
    for (int n = 0; n < DSTATE; ++n)
      h[n] = __expf(dt_t * a[n]) * h[n] + dx * bcp[DTRANK + n];
  }
  sdt[gid] = sum;
#pragma unroll
  for (int n = 0; n < DSTATE; ++n) carryS[(size_t)gid * DSTATE + n] = h[n];
}

// Pass 2: tiny sequential carry over the 64 chunks (B*DINNER threads).
// Chunk transition is exp(a[n]*sum_dt) since the products of exponentials collapse.
__global__ __launch_bounds__(256) void scan_part2(
    const float* __restrict__ carryS, const float* __restrict__ sdt,
    const float* __restrict__ Alog, float* __restrict__ Hstart) {
  const int gid = blockIdx.x * blockDim.x + threadIdx.x;  // B*DINNER
  const int d = gid & (DINNER - 1);
  const int b = gid / DINNER;
  float a[DSTATE], h[DSTATE];
#pragma unroll
  for (int n = 0; n < DSTATE; ++n) { a[n] = -__expf(Alog[d * DSTATE + n]); h[n] = 0.0f; }
  for (int c = 0; c < NCH; ++c) {
    const size_t idx = ((size_t)(b * NCH + c) * DINNER + d);
#pragma unroll
    for (int n = 0; n < DSTATE; ++n) Hstart[idx * DSTATE + n] = h[n];
    const float p = sdt[idx];
#pragma unroll
    for (int n = 0; n < DSTATE; ++n)
      h[n] = __expf(a[n] * p) * h[n] + carryS[idx * DSTATE + n];
  }
}

// Pass 3: rerun each chunk from its true start state, emit gated outputs.
__global__ __launch_bounds__(256) void scan_part3(
    const float* __restrict__ dt, const float* __restrict__ xc,
    const float* __restrict__ dbl, const float* __restrict__ xz,
    const float* __restrict__ Alog, const float* __restrict__ Dp,
    const float* __restrict__ Hstart, float* __restrict__ y,
    int reverse, int accum) {
  const int gid = blockIdx.x * blockDim.x + threadIdx.x;  // B*NCH*DINNER
  const int d = gid & (DINNER - 1);
  const int c = (gid / DINNER) & (NCH - 1);
  const int b = gid / (DINNER * NCH);
  float a[DSTATE], h[DSTATE];
#pragma unroll
  for (int n = 0; n < DSTATE; ++n) {
    a[n] = -__expf(Alog[d * DSTATE + n]);
    h[n] = Hstart[(size_t)gid * DSTATE + n];
  }
  const float dpar = Dp[d];
  for (int s = 0; s < TC; ++s) {
    const int sg = c * TC + s;
    const int t  = reverse ? (LSEQ - 1 - sg) : sg;
    const size_t base = (size_t)b * LSEQ + t;
    const float dt_t = dt[base * DINNER + d];
    const float x_t  = xc[base * DINNER + d];
    const float dx   = dt_t * x_t;
    const float* bcp = dbl + base * XPN;
    float yv = 0.0f;
#pragma unroll
    for (int n = 0; n < DSTATE; ++n) {
      h[n] = __expf(dt_t * a[n]) * h[n] + dx * bcp[DTRANK + n];
      yv += h[n] * bcp[DTRANK + DSTATE + n];
    }
    yv += dpar * x_t;
    const float z = xz[base * (2 * DINNER) + DINNER + d];
    yv *= z / (1.0f + __expf(-z));
    float* out = y + base * DINNER + d;
    if (accum) *out += yv; else *out = yv;
  }
}

// (y_f + y_r) * silu(y_g) -> bf16
__global__ __launch_bounds__(256) void combine_kernel(
    const float* __restrict__ yfr, const float* __restrict__ yg,
    bf16_t* __restrict__ out, int n) {
  int i = blockIdx.x * blockDim.x + threadIdx.x;
  if (i < n) {
    float g = yg[i];
    out[i] = (bf16_t)(yfr[i] * (g / (1.0f + __expf(-g))));
  }
}

// ---------------------------------------------------------------------------
extern "C" void kernel_launch(void* const* d_in, const int* in_sizes, int n_in,
                              void* d_out, int out_size, void* d_ws, size_t ws_size,
                              hipStream_t stream) {
  (void)in_sizes; (void)n_in; (void)out_size; (void)ws_size;
  const float* hs    = (const float*)d_in[0];
  const float* ahs   = (const float*)d_in[1];
  const float* inpw  = (const float*)d_in[2];
  const float* inpgw = (const float*)d_in[3];
  const float* convw[3] = {(const float*)d_in[4], (const float*)d_in[6], (const float*)d_in[8]};
  const float* convb[3] = {(const float*)d_in[5], (const float*)d_in[7], (const float*)d_in[9]};
  const float* xproj[3] = {(const float*)d_in[10], (const float*)d_in[11], (const float*)d_in[12]};
  const float* dtw[3]   = {(const float*)d_in[13], (const float*)d_in[15], (const float*)d_in[17]};
  const float* dtb[3]   = {(const float*)d_in[14], (const float*)d_in[16], (const float*)d_in[18]};
  const float* Alog[3]  = {(const float*)d_in[19], (const float*)d_in[20], (const float*)d_in[21]};
  const float* Dp[3]    = {(const float*)d_in[22], (const float*)d_in[23], (const float*)d_in[24]};
  const float* outpw    = (const float*)d_in[25];
  float* out = (float*)d_out;

  char* ws = (char*)d_ws;
  size_t off = 0;
  auto carve = [&](size_t bytes) -> void* {
    void* p = ws + off;
    off += (bytes + 255) & ~(size_t)255;
    return p;
  };
  bf16_t* ACTBF = (bf16_t*)carve((size_t)NROWS * DMODEL * 2);       // X / AX bf16
  float*  XZ    = (float*) carve((size_t)NROWS * 2 * DINNER * 4);   // in_proj out (reused)
  float*  XC    = (float*) carve((size_t)NROWS * DINNER * 4);       // conv+silu f32
  bf16_t* XCB   = (bf16_t*)carve((size_t)NROWS * DINNER * 2);       // conv+silu bf16
  float*  DBL   = (float*) carve((size_t)NROWS * XPN * 4);          // xproj out
  bf16_t* DTINB = (bf16_t*)carve((size_t)NROWS * DTRANK * 2);       // dbl[:, :32] bf16
  float*  DT    = (float*) carve((size_t)NROWS * DINNER * 4);       // softplus(dt)
  float*  YACC  = (float*) carve((size_t)NROWS * DINNER * 4);       // y_f + y_r
  float*  YG    = (float*) carve((size_t)NROWS * DINNER * 4);       // y_g
  bf16_t* YCB   = (bf16_t*)carve((size_t)NROWS * DINNER * 2);       // combined bf16
  float*  CARS  = (float*) carve((size_t)BSZ * NCH * DINNER * DSTATE * 4);
  float*  SDT   = (float*) carve((size_t)BSZ * NCH * DINNER * 4);
  float*  HST   = (float*) carve((size_t)BSZ * NCH * DINNER * DSTATE * 4);
  bf16_t* WBF   = (bf16_t*)carve((size_t)2 * DINNER * DMODEL * 2);  // weight staging

  const int TPB = 256;
  auto cvt = [&](const float* s, bf16_t* d, int n) {
    f32_to_bf16_kernel<<<(n + TPB - 1) / TPB, TPB, 0, stream>>>(s, d, n);
  };
  auto gemm = [&](const bf16_t* A, int lda, const bf16_t* Bt, int ldb, float* C,
                  int N, int K, const float* bias, int epi) {
    dim3 grid(N / 64, NROWS / 256);
    gemm_wmma_bf16<<<grid, 256, 0, stream>>>(A, lda, Bt, ldb, C, N, K, bias, epi);
  };

  auto run_branch = [&](int bi, int reverse, float* ybuf, int accum) {
    conv_silu_kernel<<<(NROWS * DINNER) / TPB, TPB, 0, stream>>>(
        XZ, convw[bi], convb[bi], XC, XCB, reverse);
    cvt(xproj[bi], WBF, XPN * DINNER);
    gemm(XCB, DINNER, WBF, DINNER, DBL, XPN, DINNER, nullptr, 0);
    slice_to_bf16_kernel<<<(NROWS * DTRANK) / TPB, TPB, 0, stream>>>(
        DBL, XPN, DTINB, DTRANK, DTRANK, NROWS * DTRANK);
    cvt(dtw[bi], WBF, DINNER * DTRANK);
    gemm(DTINB, DTRANK, WBF, DTRANK, DT, DINNER, DTRANK, dtb[bi], 1);
    scan_part1<<<(BSZ * NCH * DINNER) / TPB, TPB, 0, stream>>>(
        DT, XC, DBL, Alog[bi], CARS, SDT, reverse);
    scan_part2<<<(BSZ * DINNER) / TPB, TPB, 0, stream>>>(CARS, SDT, Alog[bi], HST);
    scan_part3<<<(BSZ * NCH * DINNER) / TPB, TPB, 0, stream>>>(
        DT, XC, DBL, XZ, Alog[bi], Dp[bi], HST, ybuf, reverse, accum);
  };

  // xz = hidden_states @ in_proj_w^T  (bf16 WMMA)
  cvt(hs, ACTBF, NROWS * DMODEL);
  cvt(inpw, WBF, 2 * DINNER * DMODEL);
  gemm(ACTBF, DMODEL, WBF, DMODEL, XZ, 2 * DINNER, DMODEL, nullptr, 0);

  run_branch(0, /*reverse=*/0, YACC, /*accum=*/0);   // y_f
  run_branch(1, /*reverse=*/1, YACC, /*accum=*/1);   // += y_r

  // axz = another_hidden_states @ in_proj_g_w^T (XZ buffer reused)
  cvt(ahs, ACTBF, NROWS * DMODEL);
  cvt(inpgw, WBF, 2 * DINNER * DMODEL);
  gemm(ACTBF, DMODEL, WBF, DMODEL, XZ, 2 * DINNER, DMODEL, nullptr, 0);

  run_branch(2, /*reverse=*/0, YG, /*accum=*/0);     // y_g

  combine_kernel<<<(NROWS * DINNER) / TPB, TPB, 0, stream>>>(
      YACC, YG, YCB, NROWS * DINNER);

  // out = combined @ out_proj_w^T
  cvt(outpw, WBF, DMODEL*DINNER);
  gemm(YCB, DINNER, WBF, DINNER, out, DMODEL, DINNER, nullptr, 0);
}
